// Seq2Seq_26036091748879
// MI455X (gfx1250) — matrix operand order repaired and checked by reading, not compiled, not run on hardware
//
#include <hip/hip_runtime.h>
#include <hip/hip_bf16.h>

#define BB   32      // batch
#define SS   128     // encoder seq len
#define TLEN 64      // target len
#define TDEC 63      // decoder steps (teacher forcing, :-1)
#define HID  512     // hidden = embed
#define G4   2048    // 4*H gates
#define VV   32000   // vocab

typedef unsigned short u16;
typedef __attribute__((ext_vector_type(16))) __bf16 v16bf;
typedef __attribute__((ext_vector_type(8)))  float  v8f;

__device__ __forceinline__ u16 f2bf(float x) {
  unsigned u = __float_as_uint(x);
  u += 0x7FFFu + ((u >> 16) & 1u);   // round-to-nearest-even
  return (u16)(u >> 16);
}

__device__ __forceinline__ v16bf ld16bf(const u16* p) {
  union { uint4 q[2]; v16bf v; } u;
  u.q[0] = *(const uint4*)p;
  u.q[1] = *(const uint4*)(p + 8);
  return u.v;
}
__device__ __forceinline__ v16bf ld8x2bf(const u16* p0, const u16* p1) {
  union { uint4 q[2]; v16bf v; } u;
  u.q[0] = *(const uint4*)p0;
  u.q[1] = *(const uint4*)p1;
  return u.v;
}

__device__ __forceinline__ v8f wmma_bf16(v16bf a, v16bf b, v8f c) {
  return __builtin_amdgcn_wmma_f32_16x16x32_bf16(false, a, false, b, (short)0, c,
                                                 false, false);
}

template <int EPI>
__device__ __forceinline__ void storeC(float* __restrict__ C, int N, int row,
                                       int col, float v) {
  if (EPI == 0) {
    C[(long)row * N + col] = v;
  } else {                      // row = t*32 + b  ->  out[b][t+1][col]
    const int b = row & 31;
    const int t = row >> 5;
    C[((long)(b * TLEN) + t + 1) * VV + col] = v;
  }
}

// ---------------------------------------------------------------------------
// bf16 WMMA GEMM: C[M,N] = A[M,K] * W[N,K]^T (+ bias[N])
// A row-major (leading dim lda), W row-major (ldb == K), fp32 accumulate.
// 256 threads = 8 waves. A block (32 x 128k) staged via LDS (padded stride).
// WIDE==1: block tile 32x256, each wave owns a 2x2 grid of 16x16 tiles
//          (4 WMMAs per 8 fragment loads -> 2x compute density). N%256==0.
// WIDE==0: block tile 32x64, 1 tile/wave (max parallelism for tiny M=32
//          latency-bound recurrent GEMMs). N%64==0.
// EPI==1: scatter rows m=t*32+b to out[b][t+1][n] (seq2seq logits layout).
// Requires M%32==0, K%128==0.
// ---------------------------------------------------------------------------
#define AS_LD 136

template <int EPI, int WIDE>
__global__ __launch_bounds__(256) void gemm_bf16(
    const u16* __restrict__ A, int lda, const u16* __restrict__ W,
    const float* __restrict__ bias, float* __restrict__ C,
    int M, int N, int K) {
  __shared__ u16 As[32 * AS_LD];
  const int tid  = threadIdx.x;
  const int lane = tid & 31;
  const int wave = tid >> 5;
  const int hl   = lane >> 4;
  const int l15  = lane & 15;
  const int mBlk = blockIdx.y * 32;
  // tile coordinates
  const int wm   = WIDE ? 0 : (wave >> 2);
  const int n0   = WIDE ? (blockIdx.x * 256 + wave * 32)
                        : (blockIdx.x * 64 + (wave & 3) * 16);
  const int nrow0 = n0 + l15;        // W source row, n-tile 0
  const int nrow1 = n0 + 16 + l15;   // W source row, n-tile 1 (WIDE only)
  // LDS staging assignment: 8 threads per row, 16-element segments
  const int r    = tid >> 3;
  const int cseg = (tid & 7) * 16;
  (void)M;
  v8f acc00 = {}, acc01 = {}, acc10 = {}, acc11 = {};
  for (int kc = 0; kc < K; kc += 128) {
    const u16* ga = A + (long)(mBlk + r) * lda + kc + cseg;
    uint4 q0 = *(const uint4*)ga;
    uint4 q1 = *(const uint4*)(ga + 8);
    *(uint4*)&As[r * AS_LD + cseg]     = q0;
    *(uint4*)&As[r * AS_LD + cseg + 8] = q1;
    if (kc + 128 < K) {                 // prefetch next B chunk toward L2/L0
      __builtin_prefetch(W + (long)nrow0 * K + kc + 128, 0, 1);
      if (WIDE) __builtin_prefetch(W + (long)nrow1 * K + kc + 128, 0, 1);
    }
    __syncthreads();
#pragma unroll
    for (int k0 = 0; k0 < 128; k0 += 32) {
      if (WIDE) {
        const u16* ap0 = &As[l15 * AS_LD + k0 + hl * 16];
        const u16* ap1 = ap0 + 16 * AS_LD;
        v16bf a0 = ld16bf(ap0);
        v16bf a1 = ld16bf(ap1);
        v16bf b0 = ld16bf(W + (long)nrow0 * K + kc + k0 + hl * 16);
        v16bf b1 = ld16bf(W + (long)nrow1 * K + kc + k0 + hl * 16);
        acc00 = wmma_bf16(a0, b0, acc00);
        acc01 = wmma_bf16(a0, b1, acc01);
        acc10 = wmma_bf16(a1, b0, acc10);
        acc11 = wmma_bf16(a1, b1, acc11);
      } else {
        const u16* ap = &As[(wm * 16 + l15) * AS_LD + k0 + hl * 16];
        v16bf a = ld16bf(ap);
        v16bf b = ld16bf(W + (long)nrow0 * K + kc + k0 + hl * 16);
        acc00 = wmma_bf16(a, b, acc00);
      }
    }
    __syncthreads();
  }
  if (WIDE) {
    const int colA = n0 + l15, colB = n0 + 16 + l15;
    const float bvA = bias ? bias[colA] : 0.0f;
    const float bvB = bias ? bias[colB] : 0.0f;
    const int r0 = mBlk + hl * 8;
    const int r1 = mBlk + 16 + hl * 8;
#pragma unroll
    for (int v = 0; v < 8; ++v) {
      storeC<EPI>(C, N, r0 + v, colA, acc00[v] + bvA);
      storeC<EPI>(C, N, r0 + v, colB, acc01[v] + bvB);
      storeC<EPI>(C, N, r1 + v, colA, acc10[v] + bvA);
      storeC<EPI>(C, N, r1 + v, colB, acc11[v] + bvB);
    }
  } else {
    const int col   = n0 + l15;
    const int rbase = mBlk + wm * 16 + hl * 8;
    const float bv  = bias ? bias[col] : 0.0f;
#pragma unroll
    for (int v = 0; v < 8; ++v)
      storeC<EPI>(C, N, rbase + v, col, acc00[v] + bv);
  }
}

// ---------------------------------------------------------------------------
// LSTM pointwise: gates = gx + ghh (bias baked into gx); update c,h; emit
// bf16 h for the next recurrent GEMM, optional extra fp32 stream store.
// ---------------------------------------------------------------------------
__global__ __launch_bounds__(256) void lstm_elt(
    const float* __restrict__ gx, const float* __restrict__ ghh,
    float* __restrict__ c, float* __restrict__ h,
    u16* __restrict__ hbf, int hbf_stride,
    float* __restrict__ aux, long aux_stride) {
  const int idx = blockIdx.x * 256 + threadIdx.x;
  const int b = idx >> 9, j = idx & 511;
  const int g0 = b * G4 + j;
  const float gi = gx[g0]        + ghh[g0];
  const float gf = gx[g0 + 512]  + ghh[g0 + 512];
  const float gg = gx[g0 + 1024] + ghh[g0 + 1024];
  const float go = gx[g0 + 1536] + ghh[g0 + 1536];
  const float i = 1.0f / (1.0f + __expf(-gi));
  const float f = 1.0f / (1.0f + __expf(-gf));
  const float g = tanhf(gg);
  const float o = 1.0f / (1.0f + __expf(-go));
  const int s = b * HID + j;
  const float cn = f * c[s] + i * g;
  const float hn = o * tanhf(cn);
  c[s] = cn;
  h[s] = hn;
  hbf[(long)b * hbf_stride + j] = f2bf(hn);
  if (aux) aux[(long)b * aux_stride + j] = hn;
}

// ---------------------------------------------------------------------------
// Attention (one block per batch element): scores = h_top . enc_out / sqrt(H),
// softmax over S=128, ctx = w . enc_out; emit bf16 [h_top | ctx] FC-input row.
// ---------------------------------------------------------------------------
__global__ __launch_bounds__(256) void attn_kernel(
    const float* __restrict__ h_top,    // [B][H]
    const float* __restrict__ enc_out,  // [B][S][H]
    u16* __restrict__ fcin) {           // [B][2H] (this timestep's rows)
  const int b = blockIdx.x, tid = threadIdx.x;
  __shared__ float hsm[HID];
  __shared__ float wts[SS];
  hsm[tid]       = h_top[b * HID + tid];
  hsm[tid + 256] = h_top[b * HID + 256 + tid];
  __syncthreads();
  float sc = 0.0f;
  if (tid < SS) {
    const float* e = enc_out + ((long)b * SS + tid) * HID;
    float a = 0.0f;
    for (int k = 0; k < HID; ++k) a += hsm[k] * e[k];
    sc = a * (1.0f / 22.62741699796952f);  // 1/sqrt(512)
    wts[tid] = sc;
  }
  __syncthreads();
  float mx = -3.4e38f;
  for (int s = 0; s < SS; ++s) mx = fmaxf(mx, wts[s]);
  __syncthreads();
  if (tid < SS) wts[tid] = __expf(sc - mx);
  __syncthreads();
  float sum = 0.0f;
  for (int s = 0; s < SS; ++s) sum += wts[s];
  const float inv = 1.0f / sum;
  float c0 = 0.0f, c1 = 0.0f;
  for (int s = 0; s < SS; ++s) {
    const float w = wts[s] * inv;
    const float* e = enc_out + ((long)b * SS + s) * HID;
    c0 += w * e[tid];
    c1 += w * e[tid + 256];
  }
  u16* row = fcin + (long)b * (2 * HID);
  row[tid]       = f2bf(hsm[tid]);
  row[tid + 256] = f2bf(hsm[tid + 256]);
  row[512 + tid]       = f2bf(c0);
  row[512 + 256 + tid] = f2bf(c1);
}

// ---------------------------------------------------------------------------
// Helpers: fp32->bf16 conversions, embedding gathers, fills.
// ---------------------------------------------------------------------------
__global__ void cvt_bf16(const float* __restrict__ s, u16* __restrict__ d, long n) {
  long i = (long)blockIdx.x * 256 + threadIdx.x;
  if (i < n) d[i] = f2bf(s[i]);
}
__global__ void cvt_cat(const float* __restrict__ wa, const float* __restrict__ wb,
                        u16* __restrict__ d, int N, int K) {  // d[N][2K] = [wa|wb]
  long i = (long)blockIdx.x * 256 + threadIdx.x;
  if (i >= (long)N * K) return;
  int n = (int)(i / K), k = (int)(i % K);
  d[(long)n * (2 * K) + k]     = f2bf(wa[i]);
  d[(long)n * (2 * K) + K + k] = f2bf(wb[i]);
}
__global__ void gather_emb(const int* __restrict__ seq, int Ls,
                           const float* __restrict__ emb,
                           u16* __restrict__ out, int steps) {
  long idx = (long)blockIdx.x * 256 + threadIdx.x;
  if (idx >= (long)steps * BB * HID) return;
  int j = (int)(idx & 511);
  long row = idx >> 9;
  int b = (int)(row & 31), s = (int)(row >> 5);
  int tok = seq[b * Ls + s];
  out[row * HID + j] = f2bf(emb[(long)tok * HID + j]);
}
__global__ void fill_f32(float* p, float v, long n) {
  long i = (long)blockIdx.x * 256 + threadIdx.x;
  if (i < n) p[i] = v;
}
__global__ void zero_out_t0(float* out) {  // out[b][0][:] = 0
  long i = (long)blockIdx.x * 256 + threadIdx.x;
  if (i >= (long)BB * VV) return;
  int b = (int)(i / VV), v = (int)(i % VV);
  out[(long)b * TLEN * VV + v] = 0.0f;
}

// ---------------------------------------------------------------------------
extern "C" void kernel_launch(void* const* d_in, const int* in_sizes, int n_in,
                              void* d_out, int out_size, void* d_ws, size_t ws_size,
                              hipStream_t stream) {
  (void)in_sizes; (void)n_in; (void)out_size; (void)ws_size;
  const int*   input_seq  = (const int*)d_in[0];
  const int*   target_seq = (const int*)d_in[1];
  const float* enc_emb    = (const float*)d_in[2];
  const float* dec_emb    = (const float*)d_in[3];
  const float* enc_Wih    = (const float*)d_in[4];
  const float* enc_Whh    = (const float*)d_in[5];
  const float* enc_b      = (const float*)d_in[6];
  const float* dec_Wih    = (const float*)d_in[7];
  const float* dec_Whh    = (const float*)d_in[8];
  const float* dec_b      = (const float*)d_in[9];
  const float* fc_w       = (const float*)d_in[10];
  const float* fc_b       = (const float*)d_in[11];
  float* out = (float*)d_out;

  char* ws = (char*)d_ws;
  size_t off = 0;
  auto alloc = [&](size_t bytes) -> char* {
    char* p = ws + off;
    off = (off + bytes + 255) & ~(size_t)255;
    return p;
  };

  u16* encWih_bf  = (u16*)alloc(2L * G4 * HID * 2);
  u16* encWhh_bf  = (u16*)alloc(2L * G4 * HID * 2);
  u16* decWih0_bf = (u16*)alloc((long)G4 * HID * 2);
  u16* decWhh0_bf = (u16*)alloc((long)G4 * HID * 2);
  u16* decWcat1_bf= (u16*)alloc((long)G4 * 2 * HID * 2);
  u16* fcw_bf     = (u16*)alloc((long)VV * 2 * HID * 2);
  u16* X0_bf      = (u16*)alloc((long)SS * BB * HID * 2);
  u16* X1_bf      = (u16*)alloc((long)SS * BB * HID * 2);
  float* Xpre     = (float*)alloc((long)SS * BB * G4 * 4);   // reused by decoder
  float* enc_out  = (float*)alloc((long)BB * SS * HID * 4);
  float* gates    = (float*)alloc((long)BB * G4 * 4);
  float* zgates   = (float*)alloc((long)BB * G4 * 4);
  float* state    = (float*)alloc(4L * BB * HID * 4);        // h0,h1,c0,c1
  u16* cat1_bf    = (u16*)alloc((long)BB * 2 * HID * 2);     // [h_l0 | h_l1] bf16
  u16* demb_bf    = (u16*)alloc((long)TDEC * BB * HID * 2);
  u16* FCin_bf    = (u16*)alloc((long)TDEC * BB * 2 * HID * 2);

  auto gemm = [&](const u16* A, int lda, const u16* W, const float* bias,
                  float* C, int M, int N, int K, int epi, int wide) {
    if (wide) {
      dim3 g(N / 256, M / 32);
      if (epi == 0)
        gemm_bf16<0, 1><<<g, 256, 0, stream>>>(A, lda, W, bias, C, M, N, K);
      else
        gemm_bf16<1, 1><<<g, 256, 0, stream>>>(A, lda, W, bias, C, M, N, K);
    } else {
      dim3 g(N / 64, M / 32);
      gemm_bf16<0, 0><<<g, 256, 0, stream>>>(A, lda, W, bias, C, M, N, K);
    }
  };

  // ---- init ----
  fill_f32<<<(4 * BB * HID + 255) / 256, 256, 0, stream>>>(state, 0.0f, 4L * BB * HID);
  fill_f32<<<(BB * G4 + 255) / 256, 256, 0, stream>>>(zgates, 0.0f, (long)BB * G4);
  zero_out_t0<<<((long)BB * VV + 255) / 256, 256, 0, stream>>>(out);

  // ---- weight conversions (fp32 -> bf16, row-major; decoder L1 concatenated) ----
  cvt_bf16<<<(2L * G4 * HID + 255) / 256, 256, 0, stream>>>(enc_Wih, encWih_bf, 2L * G4 * HID);
  cvt_bf16<<<(2L * G4 * HID + 255) / 256, 256, 0, stream>>>(enc_Whh, encWhh_bf, 2L * G4 * HID);
  cvt_bf16<<<((long)G4 * HID + 255) / 256, 256, 0, stream>>>(dec_Wih, decWih0_bf, (long)G4 * HID);
  cvt_bf16<<<((long)G4 * HID + 255) / 256, 256, 0, stream>>>(dec_Whh, decWhh0_bf, (long)G4 * HID);
  cvt_cat<<<((long)G4 * HID + 255) / 256, 256, 0, stream>>>(
      dec_Wih + (long)G4 * HID, dec_Whh + (long)G4 * HID, decWcat1_bf, G4, HID);
  cvt_bf16<<<((long)VV * 2 * HID + 255) / 256, 256, 0, stream>>>(fc_w, fcw_bf, (long)VV * 2 * HID);

  // ---- embedding gathers ----
  gather_emb<<<((long)SS * BB * HID + 255) / 256, 256, 0, stream>>>(
      input_seq, SS, enc_emb, X0_bf, SS);
  gather_emb<<<((long)TDEC * BB * HID + 255) / 256, 256, 0, stream>>>(
      target_seq, TLEN, dec_emb, demb_bf, TDEC);

  // ---- encoder: per layer, batch x@Wih^T+b over all timesteps, then recur ----
  u16* Xin = X0_bf;
  for (int l = 0; l < 2; ++l) {
    gemm(Xin, HID, encWih_bf + (long)l * G4 * HID, enc_b + l * G4, Xpre,
         SS * BB, G4, HID, 0, 1);
    float* hl = state + (long)l * BB * HID;
    float* cl = state + (long)(2 + l) * BB * HID;
    for (int s = 0; s < SS; ++s) {
      const float* ghh = zgates;
      if (s > 0) {
        const u16* Ah; int lda;
        if (l == 0) { Ah = X1_bf + (long)(s - 1) * BB * HID; lda = HID; }
        else        { Ah = cat1_bf + HID;                    lda = 2 * HID; }
        gemm(Ah, lda, encWhh_bf + (long)l * G4 * HID, nullptr, gates,
             BB, G4, HID, 0, 0);
        ghh = gates;
      }
      if (l == 0)
        lstm_elt<<<64, 256, 0, stream>>>(Xpre + (long)s * BB * G4, ghh, cl, hl,
                                         X1_bf + (long)s * BB * HID, HID, nullptr, 0L);
      else
        lstm_elt<<<64, 256, 0, stream>>>(Xpre + (long)s * BB * G4, ghh, cl, hl,
                                         cat1_bf + HID, 2 * HID,
                                         enc_out + (long)s * HID, (long)SS * HID);
    }
    Xin = X1_bf;
  }

  // ---- decoder: batch x@Wih0^T+b over all steps, then recur; L1 uses cat GEMM ----
  gemm(demb_bf, HID, decWih0_bf, dec_b, Xpre, TDEC * BB, G4, HID, 0, 1);
  float* h0 = state;
  float* h1 = state + (long)BB * HID;
  float* c0 = state + 2L * BB * HID;
  float* c1 = state + 3L * BB * HID;

  for (int t = 0; t < TDEC; ++t) {
    const u16* A0; int lda0;
    if (t == 0) { A0 = X1_bf + (long)(SS - 1) * BB * HID; lda0 = HID; }  // enc L0 final h
    else        { A0 = cat1_bf;                           lda0 = 2 * HID; }
    gemm(A0, lda0, decWhh0_bf, nullptr, gates, BB, G4, HID, 0, 0);
    lstm_elt<<<64, 256, 0, stream>>>(Xpre + (long)t * BB * G4, gates, c0, h0,
                                     cat1_bf, 2 * HID, nullptr, 0L);
    // layer 1: gates = [h0_new | h1_prev] @ [Wih1 | Whh1]^T + b   (K = 1024)
    gemm(cat1_bf, 2 * HID, decWcat1_bf, dec_b + G4, gates, BB, G4, 2 * HID, 0, 0);
    lstm_elt<<<64, 256, 0, stream>>>(zgates, gates, c1, h1,
                                     cat1_bf + HID, 2 * HID, nullptr, 0L);
    attn_kernel<<<BB, 256, 0, stream>>>(h1, enc_out, FCin_bf + (long)t * BB * 2 * HID);
  }

  // ---- batched FC over all 63 steps: [2016,1024] x [32000,1024]^T, L2-resident W ----
  gemm(FCin_bf, 2 * HID, fcw_bf, fc_b, out, TDEC * BB, VV, 2 * HID, 1, 1);
}
